// RegionalRefiner_61194694033516
// MI455X (gfx1250) — compile-verified, matrix operand order checked by reading, not run on hardware
//
#include <hip/hip_runtime.h>
#include <math.h>

// ---------------------------------------------------------------------------
// CDNA5 (gfx1250) windowed-critic kernel, v2.
//   conv0: 256->128 K=7 pad=3 | conv1/2: 128->128 K=5 pad=2 | mean | MLP head
// Convs as WMMA bf16 GEMMs, K tap-major so each 32-wide K block is a
// contiguous channel slice of the shifted input (im2col == shifted LDS read).
// v2 changes vs v1:
//  * +8 bf16 row padding -> LDS row stride = 4 dwords mod 64 banks:
//    B-fragment ds_load_b128 across 16 lanes is bank-conflict-free.
//  * 2 batch elements per pass: each A (weight) fragment feeds 4 WMMAs,
//    halving L2 weight streaming (~6 GB -> ~3 GB per launch).
//  * conv2 input tiles alias the dead x-stage LDS region (halo re-zeroed
//    per pass) to stay under the 64 KB static-LDS cap (63.3 KB used).
// ---------------------------------------------------------------------------

typedef __attribute__((ext_vector_type(16))) __bf16 v16bf;
typedef __attribute__((ext_vector_type(8)))  __bf16 v8bf;
typedef __attribute__((ext_vector_type(8)))  float  v8f;

#define BATCH   8
#define SEQ_T   25000
#define DIN     256
#define HID     128
#define WSZ     25
#define NWIN    1000
#define THRESH  3.5f

#define XST_ROWS   38            // conv0 reads rows t+k <= 31+6 = 37
#define XST_STRIDE (DIN + 8)     // 264 bf16 = 528 B = 33*16, 132 dw (4 mod 64)
#define XST_ELEMS  (XST_ROWS * XST_STRIDE)   // 10032
#define Y_ROWS     36            // conv1/2 read rows t+k <= 31+4 = 35
#define Y_STRIDE   (HID + 8)     // 136 bf16 = 272 B = 17*16, 68 dw (4 mod 64)
#define Y_ELEMS    (Y_ROWS * Y_STRIDE)       // 4896

// ws layout (bytes)
#define W0P_OFF 0
#define W0P_ELEMS (128*1792)                 // K = 7*256
#define W1P_OFF (W0P_OFF + W0P_ELEMS*2)      // 458752
#define W1P_ELEMS (128*640)                  // K = 5*128
#define W2P_OFF (W1P_OFF + W1P_ELEMS*2)      // 622592

__device__ __forceinline__ float gelu_exact(float x) {
    return 0.5f * x * (1.0f + erff(x * 0.7071067811865476f));
}

__device__ __forceinline__ v16bf cat8(v8bf lo, v8bf hi) {
    return __builtin_shufflevector(lo, hi, 0,1,2,3,4,5,6,7,8,9,10,11,12,13,14,15);
}

// fp32 -> bf16 weight repack, tap-major K order: Wp[c][k*Cin + d] = w[c][d][k]
__global__ void pack_weights_kernel(const float* __restrict__ w0,
                                    const float* __restrict__ w1,
                                    const float* __restrict__ w2,
                                    __bf16* __restrict__ W0p,
                                    __bf16* __restrict__ W1p,
                                    __bf16* __restrict__ W2p) {
    int idx = blockIdx.x * blockDim.x + threadIdx.x;
    const int N0 = W0P_ELEMS, N1 = W1P_ELEMS;
    if (idx < N0) {
        int c = idx / 1792, kk = idx % 1792;
        int k = kk >> 8, d = kk & 255;                       // kk = k*256 + d
        W0p[idx] = (__bf16)w0[(c * 256 + d) * 7 + k];
    } else if (idx < N0 + N1) {
        int i = idx - N0;
        int c = i / 640, kk = i % 640;
        int k = kk >> 7, d = kk & 127;                       // kk = k*128 + d
        W1p[i] = (__bf16)w1[(c * 128 + d) * 5 + k];
    } else if (idx < N0 + 2 * N1) {
        int i = idx - N0 - N1;
        int c = i / 640, kk = i % 640;
        int k = kk >> 7, d = kk & 127;
        W2p[i] = (__bf16)w2[(c * 128 + d) * 5 + k];
    }
}

__global__ __launch_bounds__(256)
void critic_windows_kernel(const float* __restrict__ latents,
                           const __bf16* __restrict__ W0p,
                           const __bf16* __restrict__ W1p,
                           const __bf16* __restrict__ W2p,
                           const float* __restrict__ b0g,
                           const float* __restrict__ b1g,
                           const float* __restrict__ b2g,
                           const float* __restrict__ hw1,
                           const float* __restrict__ hb1,
                           const float* __restrict__ hw2,
                           const float* __restrict__ hb2,
                           float* __restrict__ out) {
    // xmem: x-stage tiles for the 2 batch elements; after conv0 this region
    // is dead and y1 (conv2 input) tiles alias onto it.
    __shared__ __align__(16) __bf16 xmem [2 * XST_ELEMS];   // 40,128 B
    __shared__ __align__(16) __bf16 y0mem[2 * Y_ELEMS];     // 19,584 B
    __shared__ float featsum[2 * HID];
    __shared__ float hbuf[2 * HID];
    __shared__ float bias0[HID], bias1[HID], bias2[HID];
    __shared__ float sAcc[2];

    __bf16* const xstA = xmem;
    __bf16* const xstB = xmem + XST_ELEMS;
    __bf16* const y0A  = y0mem;
    __bf16* const y0B  = y0mem + Y_ELEMS;
    __bf16* const y1A  = xmem;               // alias: x-stage dead after conv0
    __bf16* const y1B  = xmem + Y_ELEMS;

    const int tid    = threadIdx.x;
    const int lane   = tid & 31;
    const int wave   = tid >> 5;          // 0..7 -> M tile
    const int w      = blockIdx.x;        // window 0..999
    const int m0     = wave * 16;
    const int nlo    = lane & 15;         // N within tile / A row within tile
    const int hihalf = (lane >> 4) & 1;
    const int kbase  = hihalf ? 8 : 0;    // 16-bit A/B lane->K mapping
    const int crow0  = m0 + 8 * hihalf;   // first channel of packed D column

    // one-time init: y0 halo rows must be zero (y0 region is never aliased)
    for (int i = tid; i < 2 * Y_ELEMS; i += 256) y0mem[i] = (__bf16)0.0f;
    if (tid < HID) { bias0[tid] = b0g[tid]; bias1[tid] = b1g[tid]; bias2[tid] = b2g[tid]; }
    if (tid < 2) sAcc[tid] = 0.0f;
    __syncthreads();

    for (int bp = 0; bp < BATCH / 2; ++bp) {
        // ---- stage pair of windows: xst[t+3][d] = latents[b, w*25+t, d]
        const float* srcA = latents + ((size_t)(2 * bp)     * SEQ_T + (size_t)w * WSZ) * DIN;
        const float* srcB = latents + ((size_t)(2 * bp + 1) * SEQ_T + (size_t)w * WSZ) * DIN;
        #pragma unroll 5
        for (int t = 0; t < WSZ; ++t) {
            xstA[(t + 3) * XST_STRIDE + tid] = (__bf16)srcA[t * DIN + tid];
            xstB[(t + 3) * XST_STRIDE + tid] = (__bf16)srcB[t * DIN + tid];
        }
        // re-zero x halo rows (region was clobbered by previous pass's y1)
        for (int i = tid; i < 13 * XST_STRIDE; i += 256) {
            int j = i / XST_STRIDE, col = i - j * XST_STRIDE;
            int r = (j < 3) ? j : (25 + j);          // rows 0..2, 28..37
            xstA[r * XST_STRIDE + col] = (__bf16)0.0f;
            xstB[r * XST_STRIDE + col] = (__bf16)0.0f;
        }
        featsum[tid < 256 ? tid : 0] = 0.0f;
        __syncthreads();

        // ---- conv0: K = 7*256 -> 56 blocks; src row = t + k (pad folded in)
        {
            v8f aA0 = {}, aA1 = {}, aB0 = {}, aB1 = {};
            const __bf16* wrow = W0p + (m0 + nlo) * (7 * DIN);
            for (int kb = 0; kb < 56; ++kb) {
                const int k  = kb >> 3;
                const int d0 = (kb & 7) << 5;
                const __bf16* ap = wrow + (kb << 5) + kbase;
                __builtin_prefetch(ap + 32, 0, 3);
                v16bf a = cat8(*(const v8bf*)ap, *(const v8bf*)(ap + 16));
                const int ro0 = (nlo + k) * XST_STRIDE + d0 + kbase;
                const int ro1 = (16 + nlo + k) * XST_STRIDE + d0 + kbase;
                v16bf b0 = cat8(*(const v8bf*)(xstA + ro0), *(const v8bf*)(xstA + ro0 + 16));
                aA0 = __builtin_amdgcn_wmma_f32_16x16x32_bf16(false, a, false, b0, (short)0, aA0, false, false);
                v16bf b1 = cat8(*(const v8bf*)(xstA + ro1), *(const v8bf*)(xstA + ro1 + 16));
                aA1 = __builtin_amdgcn_wmma_f32_16x16x32_bf16(false, a, false, b1, (short)0, aA1, false, false);
                v16bf b2 = cat8(*(const v8bf*)(xstB + ro0), *(const v8bf*)(xstB + ro0 + 16));
                aB0 = __builtin_amdgcn_wmma_f32_16x16x32_bf16(false, a, false, b2, (short)0, aB0, false, false);
                v16bf b3 = cat8(*(const v8bf*)(xstB + ro1), *(const v8bf*)(xstB + ro1 + 16));
                aB1 = __builtin_amdgcn_wmma_f32_16x16x32_bf16(false, a, false, b3, (short)0, aB1, false, false);
            }
            // epilogue: bias+gelu, mask t>=25 so halo rows stay zero
            v8bf oA0 = {}, oA1 = {}, oB0 = {}, oB1 = {};
            const bool t1ok = (16 + nlo < WSZ);
            #pragma unroll
            for (int r = 0; r < 8; ++r) {
                int c = crow0 + r;
                float bs = bias0[c];
                oA0[r] = (__bf16)gelu_exact(aA0[r] + bs);
                oB0[r] = (__bf16)gelu_exact(aB0[r] + bs);
                oA1[r] = (__bf16)(t1ok ? gelu_exact(aA1[r] + bs) : 0.0f);
                oB1[r] = (__bf16)(t1ok ? gelu_exact(aB1[r] + bs) : 0.0f);
            }
            *(v8bf*)(y0A + (nlo + 2)      * Y_STRIDE + crow0) = oA0;
            *(v8bf*)(y0A + (16 + nlo + 2) * Y_STRIDE + crow0) = oA1;
            *(v8bf*)(y0B + (nlo + 2)      * Y_STRIDE + crow0) = oB0;
            *(v8bf*)(y0B + (16 + nlo + 2) * Y_STRIDE + crow0) = oB1;
        }
        __syncthreads();   // x-stage region now dead -> y1 may alias it

        // zero y1 halo rows (region clobbered by x staging); conv1 epilogue
        // fills rows 2..33, so only rows 0,1,34,35 need explicit zeros.
        for (int i = tid; i < 4 * Y_STRIDE; i += 256) {
            int j = i / Y_STRIDE, col = i - j * Y_STRIDE;
            int r = (j < 2) ? j : (32 + j);          // rows 0,1,34,35
            y1A[r * Y_STRIDE + col] = (__bf16)0.0f;
            y1B[r * Y_STRIDE + col] = (__bf16)0.0f;
        }

        // ---- conv1: K = 5*128 -> 20 blocks
        {
            v8f aA0 = {}, aA1 = {}, aB0 = {}, aB1 = {};
            const __bf16* wrow = W1p + (m0 + nlo) * (5 * HID);
            for (int kb = 0; kb < 20; ++kb) {
                const int k  = kb >> 2;
                const int d0 = (kb & 3) << 5;
                const __bf16* ap = wrow + (kb << 5) + kbase;
                v16bf a = cat8(*(const v8bf*)ap, *(const v8bf*)(ap + 16));
                const int ro0 = (nlo + k) * Y_STRIDE + d0 + kbase;
                const int ro1 = (16 + nlo + k) * Y_STRIDE + d0 + kbase;
                v16bf b0 = cat8(*(const v8bf*)(y0A + ro0), *(const v8bf*)(y0A + ro0 + 16));
                aA0 = __builtin_amdgcn_wmma_f32_16x16x32_bf16(false, a, false, b0, (short)0, aA0, false, false);
                v16bf b1 = cat8(*(const v8bf*)(y0A + ro1), *(const v8bf*)(y0A + ro1 + 16));
                aA1 = __builtin_amdgcn_wmma_f32_16x16x32_bf16(false, a, false, b1, (short)0, aA1, false, false);
                v16bf b2 = cat8(*(const v8bf*)(y0B + ro0), *(const v8bf*)(y0B + ro0 + 16));
                aB0 = __builtin_amdgcn_wmma_f32_16x16x32_bf16(false, a, false, b2, (short)0, aB0, false, false);
                v16bf b3 = cat8(*(const v8bf*)(y0B + ro1), *(const v8bf*)(y0B + ro1 + 16));
                aB1 = __builtin_amdgcn_wmma_f32_16x16x32_bf16(false, a, false, b3, (short)0, aB1, false, false);
            }
            v8bf oA0 = {}, oA1 = {}, oB0 = {}, oB1 = {};
            const bool t1ok = (16 + nlo < WSZ);
            #pragma unroll
            for (int r = 0; r < 8; ++r) {
                int c = crow0 + r;
                float bs = bias1[c];
                oA0[r] = (__bf16)gelu_exact(aA0[r] + bs);
                oB0[r] = (__bf16)gelu_exact(aB0[r] + bs);
                oA1[r] = (__bf16)(t1ok ? gelu_exact(aA1[r] + bs) : 0.0f);
                oB1[r] = (__bf16)(t1ok ? gelu_exact(aB1[r] + bs) : 0.0f);
            }
            *(v8bf*)(y1A + (nlo + 2)      * Y_STRIDE + crow0) = oA0;
            *(v8bf*)(y1A + (16 + nlo + 2) * Y_STRIDE + crow0) = oA1;
            *(v8bf*)(y1B + (nlo + 2)      * Y_STRIDE + crow0) = oB0;
            *(v8bf*)(y1B + (16 + nlo + 2) * Y_STRIDE + crow0) = oB1;
        }
        __syncthreads();

        // ---- conv2 + pooled-feature accumulation
        {
            v8f aA0 = {}, aA1 = {}, aB0 = {}, aB1 = {};
            const __bf16* wrow = W2p + (m0 + nlo) * (5 * HID);
            for (int kb = 0; kb < 20; ++kb) {
                const int k  = kb >> 2;
                const int d0 = (kb & 3) << 5;
                const __bf16* ap = wrow + (kb << 5) + kbase;
                v16bf a = cat8(*(const v8bf*)ap, *(const v8bf*)(ap + 16));
                const int ro0 = (nlo + k) * Y_STRIDE + d0 + kbase;
                const int ro1 = (16 + nlo + k) * Y_STRIDE + d0 + kbase;
                v16bf b0 = cat8(*(const v8bf*)(y1A + ro0), *(const v8bf*)(y1A + ro0 + 16));
                aA0 = __builtin_amdgcn_wmma_f32_16x16x32_bf16(false, a, false, b0, (short)0, aA0, false, false);
                v16bf b1 = cat8(*(const v8bf*)(y1A + ro1), *(const v8bf*)(y1A + ro1 + 16));
                aA1 = __builtin_amdgcn_wmma_f32_16x16x32_bf16(false, a, false, b1, (short)0, aA1, false, false);
                v16bf b2 = cat8(*(const v8bf*)(y1B + ro0), *(const v8bf*)(y1B + ro0 + 16));
                aB0 = __builtin_amdgcn_wmma_f32_16x16x32_bf16(false, a, false, b2, (short)0, aB0, false, false);
                v16bf b3 = cat8(*(const v8bf*)(y1B + ro1), *(const v8bf*)(y1B + ro1 + 16));
                aB1 = __builtin_amdgcn_wmma_f32_16x16x32_bf16(false, a, false, b3, (short)0, aB1, false, false);
            }
            const bool t1ok = (16 + nlo < WSZ);
            #pragma unroll
            for (int r = 0; r < 8; ++r) {
                int c = crow0 + r;
                float bs = bias2[c];
                atomicAdd(&featsum[c],       gelu_exact(aA0[r] + bs));
                atomicAdd(&featsum[HID + c], gelu_exact(aB0[r] + bs));
                if (t1ok) {
                    atomicAdd(&featsum[c],       gelu_exact(aA1[r] + bs));
                    atomicAdd(&featsum[HID + c], gelu_exact(aB1[r] + bs));
                }
            }
        }
        __syncthreads();

        // ---- head layer 1: 128 threads per batch element
        {
            const int c = tid & 127, bb = tid >> 7;
            const float inv = 1.0f / (float)WSZ;
            float s = hb1[c];
            #pragma unroll 8
            for (int d = 0; d < HID; ++d)
                s += hw1[c * HID + d] * (featsum[bb * HID + d] * inv);
            hbuf[tid] = gelu_exact(s);
        }
        __syncthreads();

        // ---- head layer 2 + sigmoid*5
        if ((tid & 127) == 0) {
            const int bb = tid >> 7;
            float z = hb2[0];
            for (int d = 0; d < HID; ++d) z += hw2[d] * hbuf[bb * HID + d];
            sAcc[bb] += 5.0f / (1.0f + expf(-z));
        }
        __syncthreads();
    }

    if (tid == 0) {
        float sc = (sAcc[0] + sAcc[1]) * (1.0f / (float)BATCH);
        out[w]        = sc;                               // scores
        out[NWIN + w] = (sc < THRESH) ? 1.0f : 0.0f;      // low_quality flags
    }
}

extern "C" void kernel_launch(void* const* d_in, const int* in_sizes, int n_in,
                              void* d_out, int out_size, void* d_ws, size_t ws_size,
                              hipStream_t stream) {
    (void)in_sizes; (void)n_in; (void)out_size; (void)ws_size;
    const float* latents = (const float*)d_in[0];
    const float* c0w = (const float*)d_in[1];
    const float* c0b = (const float*)d_in[2];
    const float* c1w = (const float*)d_in[3];
    const float* c1b = (const float*)d_in[4];
    const float* c2w = (const float*)d_in[5];
    const float* c2b = (const float*)d_in[6];
    const float* hw1 = (const float*)d_in[7];
    const float* hb1 = (const float*)d_in[8];
    const float* hw2 = (const float*)d_in[9];
    const float* hb2 = (const float*)d_in[10];

    char* ws = (char*)d_ws;
    __bf16* W0p = (__bf16*)(ws + W0P_OFF);
    __bf16* W1p = (__bf16*)(ws + W1P_OFF);
    __bf16* W2p = (__bf16*)(ws + W2P_OFF);

    const int totalPack = W0P_ELEMS + 2 * W1P_ELEMS;   // 393,216
    pack_weights_kernel<<<(totalPack + 255) / 256, 256, 0, stream>>>(
        c0w, c1w, c2w, W0p, W1p, W2p);

    critic_windows_kernel<<<NWIN, 256, 0, stream>>>(
        latents, W0p, W1p, W2p, c0b, c1b, c2b, hw1, hb1, hw2, hb2, (float*)d_out);
}
